// FAGCN_49134425866993
// MI455X (gfx1250) — matrix operand order repaired and checked by reading, compiled, not run on hardware
//
#include <hip/hip_runtime.h>
#include <math.h>

// ---------------------------------------------------------------------------
// FAGCN + BiGRU + attention for MI455X (gfx1250, wave32, WMMA)
// ---------------------------------------------------------------------------

typedef __attribute__((ext_vector_type(16))) __bf16 v16bf;
typedef __attribute__((ext_vector_type(8)))  __bf16 v8bf;
typedef __attribute__((ext_vector_type(4)))  __bf16 v4bf;
typedef __attribute__((ext_vector_type(8)))  float  v8f;

#define HD 128          // hidden size
#define TT 3            // history length (L+1)
#define EPSC 0.3f
#define NTW 8           // 16-wide output-channel tiles per wave (128 channels)

// wave32 tree reduction (sum) across all 32 lanes
__device__ __forceinline__ float wsum(float v) {
#pragma unroll
  for (int o = 16; o > 0; o >>= 1) v += __shfl_xor(v, o, 32);
  return v;
}
__device__ __forceinline__ float sigm(float x) { return 1.f / (1.f + __expf(-x)); }

// ---------------------------------------------------------------------------
// f32 -> bf16 elementwise convert
// ---------------------------------------------------------------------------
__global__ void k_cvt(const float* __restrict__ in, __bf16* __restrict__ out, long n) {
  long i = (long)blockIdx.x * blockDim.x + threadIdx.x;
  if (i < n) out[i] = (__bf16)in[i];
}

// ---------------------------------------------------------------------------
// in-degree (exact integer atomics), then d = rsqrt(max(deg,1))
// ---------------------------------------------------------------------------
__global__ void k_deg(const int* __restrict__ dst, int* __restrict__ deg, int e) {
  int i = blockIdx.x * blockDim.x + threadIdx.x;
  if (i < e) atomicAdd(&deg[dst[i]], 1);
}
__global__ void k_dnorm(const int* __restrict__ deg, float* __restrict__ d, int n) {
  int i = blockIdx.x * blockDim.x + threadIdx.x;
  if (i < n) d[i] = rsqrtf(fmaxf((float)deg[i], 1.f));
}

// ---------------------------------------------------------------------------
// input prep: x = l2norm(h / max(sum(h),1)) -> bf16 (GEMM activation operand)
// one wave per node, lane handles 4 columns
// ---------------------------------------------------------------------------
__global__ __launch_bounds__(256)
void k_prep(const float* __restrict__ h, __bf16* __restrict__ xbf, int n) {
  int wave = threadIdx.x >> 5, lane = threadIdx.x & 31;
  int node = blockIdx.x * 8 + wave;
  if (node >= n) return;
  float4 hv = *reinterpret_cast<const float4*>(h + (long)node * HD + lane * 4);
  float s = wsum(hv.x + hv.y + hv.z + hv.w);
  float inv = 1.f / fmaxf(s, 1.f);
  float x0 = hv.x * inv, x1 = hv.y * inv, x2 = hv.z * inv, x3 = hv.w * inv;
  float nrm = sqrtf(wsum(x0 * x0 + x1 * x1 + x2 * x2 + x3 * x3));
  float r = 1.f / fmaxf(nrm, 1e-12f);
  v4bf o = { (__bf16)(x0 * r), (__bf16)(x1 * r), (__bf16)(x2 * r), (__bf16)(x3 * r) };
  *reinterpret_cast<v4bf*>(xbf + (long)node * HD + lane * 4) = o;
}

// ---------------------------------------------------------------------------
// bf16 WMMA GEMM, register-blocked:  C[node][ch] = sum_k X[node][k]*W[ch][k] + bias[ch]
//
// WMMA operand mapping (one wave = 16 nodes x 128 output channels):
//   A operand = weight tile  (M = 16 output channels, K = 32)
//       lane<16 : row ch=l16, elems = K k0..k0+15
//       lane>=16: row ch=l16, elems = K k0+16..k0+31
//       -> W stored [out][in] row-major gives contiguous v8bf loads
//   B operand = activation tile (K = 32, N = 16 nodes)
//       lane<16 : col node=l16, elems = K k0..k0+15 ; lane>=16: K k0+16..k0+31
//       -> X row-major gives contiguous v8bf loads
//   D layout : lane -> node l16 ; VGPR i -> channel half*8+i
//       -> per lane, 8 CONSECUTIVE channels of one node: two b128 stores
//
// The per-node activation row (K=128, 32 VGPRs) is loaded ONCE and reused for
// all 8 channel tiles (32 WMMAs); only the small L2-resident weight matrix is
// re-read. Activations are read once per blockIdx.y slice (gridDim.y times per GEMM).
// ---------------------------------------------------------------------------
union V16 { v16bf v; v8bf h[2]; };

__global__ __launch_bounds__(256)
void k_gemm_bf16(const __bf16* __restrict__ X, long ldx,
                 const __bf16* __restrict__ W,     // [Nout][128]
                 const float* __restrict__ bias,   // [Nout]
                 float* __restrict__ C, long ldc,
                 int Mtiles) {
  int wave = threadIdx.x >> 5, lane = threadIdx.x & 31;
  int mtile = blockIdx.x * 8 + wave;
  if (mtile >= Mtiles) return;            // whole-wave exit: EXEC stays all-1s
  int half = lane >> 4, l16 = lane & 15;
  int node = mtile * 16 + l16;
  int chTile0 = blockIdx.y * NTW;

  // B operand: this lane's node row, full K=128 resident (32 VGPRs)
  const __bf16* Xp = X + (long)node * ldx;
  V16 b[4];
#pragma unroll
  for (int k = 0; k < 4; k++) {
    b[k].h[0] = *reinterpret_cast<const v8bf*>(Xp + k * 32 + half * 16);
    b[k].h[1] = *reinterpret_cast<const v8bf*>(Xp + k * 32 + half * 16 + 8);
  }

  v8f zero = {};
  v8f acc[NTW];
#pragma unroll
  for (int nt = 0; nt < NTW; nt++) acc[nt] = zero;

  const __bf16* Wp = W + (long)(chTile0 * 16 + l16) * HD;
#pragma unroll
  for (int nt = 0; nt < NTW; nt++) {
    const __bf16* Wrow = Wp + (long)nt * 16 * HD;
#pragma unroll
    for (int k = 0; k < 4; k++) {
      V16 a;
      a.h[0] = *reinterpret_cast<const v8bf*>(Wrow + k * 32 + half * 16);
      a.h[1] = *reinterpret_cast<const v8bf*>(Wrow + k * 32 + half * 16 + 8);
      acc[nt] = __builtin_amdgcn_wmma_f32_16x16x32_bf16(false, a.v, false, b[k].v,
                                                        (short)0, acc[nt], false, false);
    }
  }

  // epilogue: two contiguous b128 stores per channel tile
  float* Crow = C + (long)node * ldc;
#pragma unroll
  for (int nt = 0; nt < NTW; nt++) {
    int ch = (chTile0 + nt) * 16 + half * 8;
    float4 bv0 = *reinterpret_cast<const float4*>(bias + ch);
    float4 bv1 = *reinterpret_cast<const float4*>(bias + ch + 4);
    float4 o0 = { acc[nt][0] + bv0.x, acc[nt][1] + bv0.y,
                  acc[nt][2] + bv0.z, acc[nt][3] + bv0.w };
    float4 o1 = { acc[nt][4] + bv1.x, acc[nt][5] + bv1.y,
                  acc[nt][6] + bv1.z, acc[nt][7] + bv1.w };
    *reinterpret_cast<float4*>(Crow + ch) = o0;
    *reinterpret_cast<float4*>(Crow + ch + 4) = o1;
  }
}

// ---------------------------------------------------------------------------
// GraphNorm (contiguous equal segments) + SELU. One block per graph.
// 512 threads: col = tid&127, 4 row-groups reduced through LDS.
// ---------------------------------------------------------------------------
__global__ __launch_bounds__(512)
void k_gnorm_selu(const float* __restrict__ x, const float* __restrict__ w,
                  const float* __restrict__ b, const float* __restrict__ ms,
                  float* __restrict__ h1, int npg) {
  __shared__ float red[4][HD];
  int g = blockIdx.x;
  int c = threadIdx.x & (HD - 1);
  int rg = threadIdx.x >> 7;
  const float* xg = x + (long)g * npg * HD;
  float acc = 0.f;
  for (int r = rg; r < npg; r += 4) acc += xg[(long)r * HD + c];
  red[rg][c] = acc;
  __syncthreads();
  float mean = (red[0][c] + red[1][c] + red[2][c] + red[3][c]) / (float)npg;
  float msub = mean * ms[c];
  __syncthreads();
  acc = 0.f;
  for (int r = rg; r < npg; r += 4) {
    float s = xg[(long)r * HD + c] - msub;
    acc += s * s;
  }
  red[rg][c] = acc;
  __syncthreads();
  float var = (red[0][c] + red[1][c] + red[2][c] + red[3][c]) / (float)npg;
  float istd = rsqrtf(var + 1e-6f);
  float wc = w[c], bc = b[c];
  float* hg = h1 + (long)g * npg * HD;
  const float kScale = 1.0507009873554805f, kAlpha = 1.6732632423543772f;
  for (int r = rg; r < npg; r += 4) {
    float s = xg[(long)r * HD + c] - msub;
    float v = wc * s * istd + bc;
    hg[(long)r * HD + c] = v > 0.f ? kScale * v : kScale * kAlpha * (__expf(v) - 1.f);
  }
}

// ---------------------------------------------------------------------------
// per-node gate projections: gd = h1 . w[:128], gs = h1 . w[128:]
// ---------------------------------------------------------------------------
__global__ __launch_bounds__(256)
void k_gates(const float* __restrict__ h1, const float* __restrict__ gw,
             float* __restrict__ gd, float* __restrict__ gs, int n) {
  int wave = threadIdx.x >> 5, lane = threadIdx.x & 31;
  int node = blockIdx.x * 8 + wave;
  if (node >= n) return;
  float4 hv = *reinterpret_cast<const float4*>(h1 + (long)node * HD + lane * 4);
  float4 wd = *reinterpret_cast<const float4*>(gw + lane * 4);
  float4 wv = *reinterpret_cast<const float4*>(gw + HD + lane * 4);
  float dd = wsum(hv.x * wd.x + hv.y * wd.y + hv.z * wd.z + hv.w * wd.w);
  float ss = wsum(hv.x * wv.x + hv.y * wv.y + hv.z * wv.z + hv.w * wv.w);
  if (lane == 0) { gd[node] = dd; gs[node] = ss; }
}

// ---------------------------------------------------------------------------
// edge SpMM: z[dst] += h1[src] * tanh(gd[dst]+gs[src]+b)*d[dst]*d[src]
// one wave per edge; native f32 L2 atomics (z fits in 192MB L2)
// ---------------------------------------------------------------------------
__global__ __launch_bounds__(256)
void k_edge(const int* __restrict__ src, const int* __restrict__ dst,
            const float* __restrict__ h1, const float* __restrict__ gd,
            const float* __restrict__ gs, const float* __restrict__ dn,
            const float* __restrict__ gate_b, int layer,
            float* __restrict__ z, int ne) {
  int wave = threadIdx.x >> 5, lane = threadIdx.x & 31;
  int e = blockIdx.x * 8 + wave;
  if (e >= ne) return;
  int s = src[e], t = dst[e];
  float coef = tanhf(gd[t] + gs[s] + gate_b[layer]) * dn[t] * dn[s];
  float4 hv = *reinterpret_cast<const float4*>(h1 + (long)s * HD + lane * 4);
  float* zp = z + (long)t * HD + lane * 4;
  __hip_atomic_fetch_add(zp + 0, hv.x * coef, __ATOMIC_RELAXED, __HIP_MEMORY_SCOPE_AGENT);
  __hip_atomic_fetch_add(zp + 1, hv.y * coef, __ATOMIC_RELAXED, __HIP_MEMORY_SCOPE_AGENT);
  __hip_atomic_fetch_add(zp + 2, hv.z * coef, __ATOMIC_RELAXED, __HIP_MEMORY_SCOPE_AGENT);
  __hip_atomic_fetch_add(zp + 3, hv.w * coef, __ATOMIC_RELAXED, __HIP_MEMORY_SCOPE_AGENT);
}

// ---------------------------------------------------------------------------
// x_{l+1} = l2norm(EPS*raw + x + l2norm(z)*||x||*msg_scale); write f32 + bf16
// ---------------------------------------------------------------------------
__global__ __launch_bounds__(256)
void k_update(const float* __restrict__ z, const float* __restrict__ x,
              const float* __restrict__ raw, const float* __restrict__ msc, int layer,
              float* __restrict__ xo, __bf16* __restrict__ xobf, int n) {
  int wave = threadIdx.x >> 5, lane = threadIdx.x & 31;
  int node = blockIdx.x * 8 + wave;
  if (node >= n) return;
  long base = (long)node * HD + lane * 4;
  float4 zv = *reinterpret_cast<const float4*>(z + base);
  float4 xv = *reinterpret_cast<const float4*>(x + base);
  float4 rv = *reinterpret_cast<const float4*>(raw + base);
  float zn = sqrtf(wsum(zv.x * zv.x + zv.y * zv.y + zv.z * zv.z + zv.w * zv.w));
  float xn = sqrtf(wsum(xv.x * xv.x + xv.y * xv.y + xv.z * xv.z + xv.w * xv.w));
  float f = (xn / fmaxf(zn, 1e-12f)) * msc[layer];
  float o0 = EPSC * rv.x + xv.x + zv.x * f;
  float o1 = EPSC * rv.y + xv.y + zv.y * f;
  float o2 = EPSC * rv.z + xv.z + zv.z * f;
  float o3 = EPSC * rv.w + xv.w + zv.w * f;
  float on = sqrtf(wsum(o0 * o0 + o1 * o1 + o2 * o2 + o3 * o3));
  float r = 1.f / fmaxf(on, 1e-12f);
  o0 *= r; o1 *= r; o2 *= r; o3 *= r;
  float4 of = { o0, o1, o2, o3 };
  *reinterpret_cast<float4*>(xo + base) = of;
  v4bf ob = { (__bf16)o0, (__bf16)o1, (__bf16)o2, (__bf16)o3 };
  *reinterpret_cast<v4bf*>(xobf + base) = ob;
}

// ---------------------------------------------------------------------------
// GRU cell + attention score contribution for this direction
// ---------------------------------------------------------------------------
__global__ __launch_bounds__(256)
void k_gru_cell(const float* __restrict__ gi, const float* __restrict__ gh,
                float* __restrict__ hs, __bf16* __restrict__ hsbf,
                const float* __restrict__ attw,   // 128 floats for this direction
                float* __restrict__ scores, int t, int n) {
  int wave = threadIdx.x >> 5, lane = threadIdx.x & 31;
  int node = blockIdx.x * 8 + wave;
  if (node >= n) return;
  long b3 = (long)node * 384 + lane * 4;
  long b1 = (long)node * HD + lane * 4;
  float4 ir = *reinterpret_cast<const float4*>(gi + b3);
  float4 iz = *reinterpret_cast<const float4*>(gi + b3 + HD);
  float4 in_ = *reinterpret_cast<const float4*>(gi + b3 + 2 * HD);
  float4 hr = *reinterpret_cast<const float4*>(gh + b3);
  float4 hz = *reinterpret_cast<const float4*>(gh + b3 + HD);
  float4 hn = *reinterpret_cast<const float4*>(gh + b3 + 2 * HD);
  float4 hp = *reinterpret_cast<const float4*>(hs + b1);
  float r0 = sigm(ir.x + hr.x), r1 = sigm(ir.y + hr.y), r2 = sigm(ir.z + hr.z), r3 = sigm(ir.w + hr.w);
  float z0 = sigm(iz.x + hz.x), z1 = sigm(iz.y + hz.y), z2 = sigm(iz.z + hz.z), z3 = sigm(iz.w + hz.w);
  float n0 = tanhf(in_.x + r0 * hn.x), n1 = tanhf(in_.y + r1 * hn.y);
  float n2 = tanhf(in_.z + r2 * hn.z), n3 = tanhf(in_.w + r3 * hn.w);
  float h0 = (1.f - z0) * n0 + z0 * hp.x;
  float h1v = (1.f - z1) * n1 + z1 * hp.y;
  float h2 = (1.f - z2) * n2 + z2 * hp.z;
  float h3 = (1.f - z3) * n3 + z3 * hp.w;
  float4 ho = { h0, h1v, h2, h3 };
  *reinterpret_cast<float4*>(hs + b1) = ho;
  v4bf hb = { (__bf16)h0, (__bf16)h1v, (__bf16)h2, (__bf16)h3 };
  *reinterpret_cast<v4bf*>(hsbf + b1) = hb;
  float4 aw = *reinterpret_cast<const float4*>(attw + lane * 4);
  float dot = wsum(h0 * aw.x + h1v * aw.y + h2 * aw.z + h3 * aw.w);
  if (lane == 0) scores[(long)node * TT + t] += dot;
}

// ---------------------------------------------------------------------------
// softmax over T, weighted sum of history, final l2norm
// ---------------------------------------------------------------------------
__global__ __launch_bounds__(256)
void k_out(const float* __restrict__ hist, const float* __restrict__ scores,
           const float* __restrict__ attb, float* __restrict__ out, int n) {
  int wave = threadIdx.x >> 5, lane = threadIdx.x & 31;
  int node = blockIdx.x * 8 + wave;
  if (node >= n) return;
  float s0 = scores[(long)node * TT + 0] + attb[0];
  float s1 = scores[(long)node * TT + 1] + attb[0];
  float s2 = scores[(long)node * TT + 2] + attb[0];
  float m = fmaxf(s0, fmaxf(s1, s2));
  float e0 = __expf(s0 - m), e1 = __expf(s1 - m), e2 = __expf(s2 - m);
  float inv = 1.f / (e0 + e1 + e2);
  float a0 = e0 * inv, a1 = e1 * inv, a2 = e2 * inv;
  long b = (long)node * HD + lane * 4;
  long slot = (long)n * HD;
  float4 x0 = *reinterpret_cast<const float4*>(hist + b);
  float4 x1 = *reinterpret_cast<const float4*>(hist + slot + b);
  float4 x2 = *reinterpret_cast<const float4*>(hist + 2 * slot + b);
  float o0 = a0 * x0.x + a1 * x1.x + a2 * x2.x;
  float o1 = a0 * x0.y + a1 * x1.y + a2 * x2.y;
  float o2 = a0 * x0.z + a1 * x1.z + a2 * x2.z;
  float o3 = a0 * x0.w + a1 * x1.w + a2 * x2.w;
  float on = sqrtf(wsum(o0 * o0 + o1 * o1 + o2 * o2 + o3 * o3));
  float r = 1.f / fmaxf(on, 1e-12f);
  float4 of = { o0 * r, o1 * r, o2 * r, o3 * r };
  *reinterpret_cast<float4*>(out + b) = of;
}

// ---------------------------------------------------------------------------
// host orchestration
// ---------------------------------------------------------------------------
extern "C" void kernel_launch(void* const* d_in, const int* in_sizes, int n_in,
                              void* d_out, int out_size, void* d_ws, size_t ws_size,
                              hipStream_t stream) {
  (void)n_in; (void)out_size; (void)ws_size;
  const float* h      = (const float*)d_in[0];
  const float* t1_w   = (const float*)d_in[1];
  const float* t1_b   = (const float*)d_in[2];
  const float* gate_w = (const float*)d_in[3];
  const float* gate_b = (const float*)d_in[4];
  const float* gn_w   = (const float*)d_in[5];
  const float* gn_b   = (const float*)d_in[6];
  const float* gn_ms  = (const float*)d_in[7];
  const float* msg_sc = (const float*)d_in[8];
  const float* w_ih   = (const float*)d_in[9];
  const float* w_hh   = (const float*)d_in[10];
  const float* b_ih   = (const float*)d_in[11];
  const float* b_hh   = (const float*)d_in[12];
  const float* att_w  = (const float*)d_in[13];
  const float* att_b  = (const float*)d_in[14];
  const int*   srcI   = (const int*)d_in[15];
  const int*   dstI   = (const int*)d_in[16];

  const int N = in_sizes[0] / HD;
  const int E = in_sizes[15];
  const int B = in_sizes[17];
  const int npg = N / B;

  // --- carve workspace (256B aligned) ---
  char* ws = (char*)d_ws;
  size_t off = 0;
  auto take = [&](size_t bytes) -> char* {
    char* p = ws + off;
    off += (bytes + 255) & ~(size_t)255;
    return p;
  };
  float*  hist32 = (float*) take((size_t)TT * N * HD * 4);
  __bf16* histbf = (__bf16*)take((size_t)TT * N * HD * 2);
  __bf16* xpbf   = (__bf16*)take((size_t)N * HD * 2);
  float*  h1buf  = (float*) take((size_t)N * HD * 4);
  float*  zbuf   = (float*) take((size_t)N * HD * 4);
  float*  gibuf  = (float*) take((size_t)N * 384 * 4);
  float*  ghbuf  = (float*) take((size_t)N * 384 * 4);
  float*  hst    = (float*) take((size_t)N * HD * 4);
  __bf16* hstbf  = (__bf16*)take((size_t)N * HD * 2);
  float*  gdb    = (float*) take((size_t)N * 4);
  float*  gsb    = (float*) take((size_t)N * 4);
  float*  dnorm  = (float*) take((size_t)N * 4);
  int*    degi   = (int*)   take((size_t)N * 4);
  float*  scores = (float*) take((size_t)N * TT * 4);
  __bf16* t1wbf  = (__bf16*)take((size_t)HD * HD * 2);
  __bf16* wihbf  = (__bf16*)take((size_t)2 * 384 * HD * 2);
  __bf16* whhbf  = (__bf16*)take((size_t)2 * 384 * HD * 2);

  const int nodeBlocks = (N + 7) / 8;    // one wave (32 lanes) per node
  const int Mtiles = N / 16;             // N = 100000 -> 6250 exact

  // weights -> bf16 (once per launch)
  k_cvt<<<(HD * HD + 255) / 256, 256, 0, stream>>>(t1_w, t1wbf, (long)HD * HD);
  k_cvt<<<(2 * 384 * HD + 255) / 256, 256, 0, stream>>>(w_ih, wihbf, (long)2 * 384 * HD);
  k_cvt<<<(2 * 384 * HD + 255) / 256, 256, 0, stream>>>(w_hh, whhbf, (long)2 * 384 * HD);

  // degree norm
  hipMemsetAsync(degi, 0, (size_t)N * 4, stream);
  k_deg<<<(E + 255) / 256, 256, 0, stream>>>(dstI, degi, E);
  k_dnorm<<<(N + 255) / 256, 256, 0, stream>>>(degi, dnorm, N);

  // input transform + t1 GEMM -> hist slot 0 ("raw")
  k_prep<<<nodeBlocks, 256, 0, stream>>>(h, xpbf, N);
  dim3 g1((Mtiles + 7) / 8, HD / (16 * NTW));      // y = 1
  k_gemm_bf16<<<g1, 256, 0, stream>>>(xpbf, HD, t1wbf, t1_b, hist32, HD, Mtiles);
  {
    long nel = (long)N * HD;
    k_cvt<<<(unsigned)((nel + 255) / 256), 256, 0, stream>>>(hist32, histbf, nel);
  }

  // FAGCN layers
  for (int l = 0; l < 2; l++) {
    float* xcur = hist32 + (size_t)l * N * HD;
    k_gnorm_selu<<<B, 512, 0, stream>>>(xcur, gn_w + l * HD, gn_b + l * HD,
                                        gn_ms + l * HD, h1buf, npg);
    k_gates<<<nodeBlocks, 256, 0, stream>>>(h1buf, gate_w + l * 2 * HD, gdb, gsb, N);
    hipMemsetAsync(zbuf, 0, (size_t)N * HD * 4, stream);
    k_edge<<<(E + 7) / 8, 256, 0, stream>>>(srcI, dstI, h1buf, gdb, gsb, dnorm,
                                            gate_b, l, zbuf, E);
    k_update<<<nodeBlocks, 256, 0, stream>>>(zbuf, xcur, hist32, msg_sc, l,
                                             hist32 + (size_t)(l + 1) * N * HD,
                                             histbf + (size_t)(l + 1) * N * HD, N);
  }

  // bidirectional GRU + attention scores
  hipMemsetAsync(scores, 0, (size_t)N * TT * 4, stream);
  dim3 gg((Mtiles + 7) / 8, 384 / (16 * NTW));     // y = 3
  for (int dir = 0; dir < 2; dir++) {
    hipMemsetAsync(hst, 0, (size_t)N * HD * 4, stream);
    hipMemsetAsync(hstbf, 0, (size_t)N * HD * 2, stream);
    for (int s = 0; s < TT; s++) {
      int t = (dir == 0) ? s : (TT - 1 - s);
      k_gemm_bf16<<<gg, 256, 0, stream>>>(histbf + (size_t)t * N * HD, HD,
                                          wihbf + (size_t)dir * 384 * HD,
                                          b_ih + dir * 384, gibuf, 384, Mtiles);
      k_gemm_bf16<<<gg, 256, 0, stream>>>(hstbf, HD,
                                          whhbf + (size_t)dir * 384 * HD,
                                          b_hh + dir * 384, ghbuf, 384, Mtiles);
      k_gru_cell<<<nodeBlocks, 256, 0, stream>>>(gibuf, ghbuf, hst, hstbf,
                                                 att_w + dir * HD, scores, t, N);
    }
  }

  // softmax over T, weighted history sum, final l2norm -> d_out
  k_out<<<nodeBlocks, 256, 0, stream>>>(hist32, scores, att_b, (float*)d_out, N);
}
